// GATWalkerAgent_72095321030713
// MI455X (gfx1250) — matrix operand order repaired and checked by reading, not compiled, not run on hardware
//
#include <hip/hip_runtime.h>

#define ALPHA 0.2f

typedef __attribute__((ext_vector_type(16))) _Float16 v16h;
typedef __attribute__((ext_vector_type(8)))  _Float16 v8h;
typedef __attribute__((ext_vector_type(4)))  _Float16 v4h;
typedef __attribute__((ext_vector_type(2)))  _Float16 v2h;
typedef __attribute__((ext_vector_type(8)))  float    v8f;
typedef __attribute__((ext_vector_type(4)))  unsigned u32x4;
typedef __attribute__((ext_vector_type(8)))  int      i32x8;
typedef __attribute__((ext_vector_type(4)))  int      i32x4;

#if __has_builtin(__builtin_amdgcn_tensor_load_to_lds) && __has_builtin(__builtin_amdgcn_s_wait_tensorcnt)
#define ATHENA_HAS_TDM 1
#else
#define ATHENA_HAS_TDM 0
#endif

// ---------- order-preserving float <-> uint encoding for atomic segment-max ----------
__device__ __forceinline__ unsigned enc_f32(float f) {
    unsigned u = __float_as_uint(f);
    return (u & 0x80000000u) ? ~u : (u | 0x80000000u);
}
__device__ __forceinline__ float dec_f32(unsigned k) {
    unsigned u = (k & 0x80000000u) ? (k & 0x7FFFFFFFu) : ~k;
    return __uint_as_float(u);
}
#define ENC_NEG_INF 0x007FFFFFu   // enc(-inf)

__device__ __forceinline__ float elu_f(float x) {
    return x > 0.f ? x : (__expf(x) - 1.f);
}

// ============================ utility kernels ============================

__global__ void fill_u32(unsigned* __restrict__ p, unsigned v, long long n) {
    long long i = (long long)blockIdx.x * blockDim.x + threadIdx.x;
    if (i < n) p[i] = v;
}

__global__ void f32_to_f16_k(const float* __restrict__ in, _Float16* __restrict__ out, long long n) {
    long long i = (long long)blockIdx.x * blockDim.x + threadIdx.x;
    if (i < n) out[i] = (_Float16)in[i];
}

// W_heads [8][256][64] -> Wcat [256][512] (heads concatenated along columns), f16
__global__ void make_wcat_k(const float* __restrict__ Wh, _Float16* __restrict__ out) {
    long long i = (long long)blockIdx.x * blockDim.x + threadIdx.x;   // over 8*256*64
    if (i >= 8LL * 256 * 64) return;
    int j = (int)(i % 64);
    int k = (int)((i / 64) % 256);
    int h = (int)(i / (64 * 256));
    out[(size_t)k * 512 + h * 64 + j] = (_Float16)Wh[i];
}

// ============================ WMMA GEMM ============================
// C[M,N](f32 or f16) = act(A[M,K](f16,rowmaj) @ B[K,N](f16,rowmaj) + bias)
// Block: 256 thr = 8 waves; block tile 32x64; wave tile 16x16; K step 32.
// A tile staged global->LDS by the Tensor Data Mover (wave 0 issues, padding
// gives the 40-half LDS row stride, tensor-dim clipping zero-fills the M edge).
template<int OUT_F16, int HAS_BIAS, int DO_ELU>
__global__ __launch_bounds__(256) void gemm_wmma_f16(
    const _Float16* __restrict__ A, const _Float16* __restrict__ B,
    void* __restrict__ C, const float* __restrict__ bias,
    int M, int N, int K)
{
    __shared__ __align__(16) _Float16 As[32 * 40];   // 32 rows, stride 40 halfs
    __shared__ __align__(16) _Float16 Bt[64 * 40];   // B transposed: [n][k], stride 40

    const int tid  = threadIdx.x;
    const int lane = tid & 31;
    const int wave = tid >> 5;
    const int hlf  = lane >> 4;       // half-wave select
    const int l15  = lane & 15;
    const int mt   = wave & 1;        // 0..1 row-tile of wave
    const int nt   = wave >> 1;       // 0..3 col-tile of wave
    const int m0   = blockIdx.y * 32;
    const int n0   = blockIdx.x * 64;

    const int br = (tid * 8) >> 6, bc = (tid * 8) & 63;   // B staging: 8 halfs/thread
#if !ATHENA_HAS_TDM
    const int ar = (tid * 4) >> 5, ac = (tid * 4) & 31;   // A staging fallback
#endif

    v8f acc = {};
    for (int k0 = 0; k0 < K; k0 += 32) {
#if ATHENA_HAS_TDM
        // ---- A tile 32x32 via Tensor Data Mover (one wave issues the DMA) ----
        if (wave == 0) {
            const unsigned lds_base = (unsigned)((size_t)(&As[0]) & 0xFFFFFFFFu);
            const unsigned long long ga =
                (unsigned long long)(size_t)(A + (size_t)m0 * K + k0);
            u32x4 g0;
            g0[0] = 1u;                                       // count=1 (valid user D#)
            g0[1] = lds_base;                                 // lds_addr
            g0[2] = (unsigned)(ga & 0xFFFFFFFFu);             // global_addr[31:0]
            g0[3] = (unsigned)((ga >> 32) & 0x01FFFFFFu) | (2u << 30);  // [56:32] | type=2
            const unsigned td0 = (unsigned)(K - k0);          // remaining cols (>=32)
            const unsigned td1 = (unsigned)(M - m0);          // remaining rows (clips edge)
            i32x8 g1;
            g1[0] = (int)((1u << 16)      // data_size = 2 bytes
                        | (1u << 20)      // pad_enable
                        | (3u << 22)      // pad_interval = 16 DWORDs (= 32-half row)
                        | (3u << 25));    // pad_amount   = 4 DWORDs  (row stride 40 halfs)
            g1[1] = (int)((td0 & 0xFFFFu) << 16);                             // tensor_dim0 lo
            g1[2] = (int)(((td0 >> 16) & 0xFFFFu) | ((td1 & 0xFFFFu) << 16)); // dim0 hi | dim1 lo
            g1[3] = (int)(((td1 >> 16) & 0xFFFFu) | (32u << 16));             // dim1 hi | tile_dim0=32
            g1[4] = 32;                                                       // tile_dim1=32
            g1[5] = (int)(unsigned)K;                                         // tensor_dim0_stride
            g1[6] = 0; g1[7] = 0;
            i32x4 z4 = {};
#if __has_include(<hip/amd_detail/amd_gfx1250_TDM.h>)
            i32x8 z8 = {};
            __builtin_amdgcn_tensor_load_to_lds(g0, g1, z4, z4, z8, 0);
#else
            __builtin_amdgcn_tensor_load_to_lds(g0, g1, z4, z4, 0);
#endif
            __builtin_amdgcn_s_wait_tensorcnt(0);
        }
#else
        // ---- A tile 32x32, manual staging (guard M edge) ----
        v4h av = {};
        const int gm = m0 + ar;
        if (gm < M) av = *(const v4h*)(A + (size_t)gm * K + (k0 + ac));
        *(v4h*)(&As[ar * 40 + ac]) = av;
#endif
        // ---- B tile 32x64, transposed into LDS ----
        v8h bv = *(const v8h*)(B + (size_t)(k0 + br) * N + (n0 + bc));
        #pragma unroll
        for (int j = 0; j < 8; ++j) Bt[(bc + j) * 40 + br] = bv[j];
        __syncthreads();

        // A fragment: row m = l15, K pairs at {2j | 16+2(j-4)} + hlf*8  (ISA 7.12.2)
        v16h af;
        const int arow = (mt * 16 + l15) * 40;
        #pragma unroll
        for (int j = 0; j < 8; ++j) {
            const int kk = (j < 4 ? 2 * j : 8 + 2 * j) + hlf * 8;
            v2h p = *(const v2h*)(&As[arow + kk]);
            af[2 * j] = p[0]; af[2 * j + 1] = p[1];
        }
        // B fragment: lane column n = l15, K = hlf*8 + e (e<8), 16 + hlf*8 + (e-8) (e>=8)
        const int boff = (nt * 16 + l15) * 40 + hlf * 8;
        v8h b0 = *(const v8h*)(&Bt[boff]);
        v8h b1 = *(const v8h*)(&Bt[boff + 16]);
        v16h bf = __builtin_shufflevector(b0, b1, 0,1,2,3,4,5,6,7,8,9,10,11,12,13,14,15);

        acc = __builtin_amdgcn_wmma_f32_16x16x32_f16(
            /*neg_a=*/false, af, /*neg_b=*/false, bf,
            /*c_mod=*/(short)0, acc, /*reuse_a=*/false, /*reuse_b=*/false);
        __syncthreads();
    }

    // epilogue: C layout VGPR r -> M = r + 8*hlf, N = l15 (per 16x16 tile)
    const int gn = n0 + nt * 16 + l15;
    float bsv = HAS_BIAS ? bias[gn] : 0.f;
    #pragma unroll
    for (int r = 0; r < 8; ++r) {
        const int gm2 = m0 + mt * 16 + hlf * 8 + r;
        if (gm2 < M) {
            float v = acc[r];
            if (HAS_BIAS) v += bsv;
            if (DO_ELU)   v = elu_f(v);
            if (OUT_F16) ((_Float16*)C)[(size_t)gm2 * N + gn] = (_Float16)v;
            else         ((float*)C)[(size_t)gm2 * N + gn]    = v;
        }
    }
}

// ============================ attention kernels ============================

// layer 1: p_dst/p_src [8][N] from H16 [N][512] and a_heads [8][128]
__global__ void head_proj1(const _Float16* __restrict__ H16, const float* __restrict__ a,
                           float* __restrict__ pd, float* __restrict__ ps, int NV) {
    long long i = (long long)blockIdx.x * blockDim.x + threadIdx.x;   // i = h*NV + n
    if (i >= 8LL * NV) return;
    int h = (int)(i / NV), n = (int)(i % NV);
    const _Float16* hv = H16 + (size_t)n * 512 + h * 64;
    const float* ah = a + h * 128;
    float d = 0.f, s = 0.f;
    #pragma unroll 8
    for (int f = 0; f < 64; ++f) { float x = (float)hv[f]; d += x * ah[f]; s += x * ah[64 + f]; }
    pd[i] = d; ps[i] = s;
}

// layer 2: p2 [N] from H2 [N][256] and a_out [512]
__global__ void head_proj2(const _Float16* __restrict__ H2, const float* __restrict__ a,
                           float* __restrict__ pd, float* __restrict__ ps, int NV) {
    long long n = (long long)blockIdx.x * blockDim.x + threadIdx.x;
    if (n >= NV) return;
    const _Float16* hv = H2 + (size_t)n * 256;
    float d = 0.f, s = 0.f;
    #pragma unroll 8
    for (int f = 0; f < 256; ++f) { float x = (float)hv[f]; d += x * a[f]; s += x * a[256 + f]; }
    pd[n] = d; ps[n] = s;
}

__global__ void edge_max1(const int* __restrict__ dst, const int* __restrict__ src,
                          const float* __restrict__ pd, const float* __restrict__ ps,
                          unsigned* __restrict__ menc, long long E, int NV) {
    long long i = (long long)blockIdx.x * blockDim.x + threadIdx.x;
    if (i >= E * 8) return;
    long long e = i >> 3; int h = (int)(i & 7);
    int d = dst[e], s = src[e];
    size_t hb = (size_t)h * NV;
    float z = pd[hb + d] + ps[hb + s];
    float sc = -(z > 0.f ? z : ALPHA * z);
    atomicMax(&menc[hb + d], enc_f32(sc));
}

// one wave32 per edge: lanes 0..7 hold e for heads 0..7, shfl-broadcast per feature group
__global__ __launch_bounds__(256) void edge_acc1(
    const int* __restrict__ dst, const int* __restrict__ src,
    const float* __restrict__ pd, const float* __restrict__ ps,
    const unsigned* __restrict__ menc, float* __restrict__ den,
    const _Float16* __restrict__ H16, float* __restrict__ msg,
    long long E, int NV)
{
    const int lane = threadIdx.x & 31;
    long long e = (long long)blockIdx.x * 8 + (threadIdx.x >> 5);
    if (e >= E) return;
    const int d = dst[e], s = src[e];
    const int h = lane & 7;
    const size_t hb = (size_t)h * NV;
    float z = pd[hb + d] + ps[hb + s];
    float sc = -(z > 0.f ? z : ALPHA * z);
    float ev = __expf(sc - dec_f32(menc[hb + d]));
    if (lane < 8) atomicAdd(&den[hb + d], ev);
    const _Float16* hs = H16 + (size_t)s * 512;
    float* md = msg + (size_t)d * 512;
    __builtin_prefetch(hs, 0, 1);
    #pragma unroll
    for (int t = 0; t < 16; ++t) {
        const int f = lane + 32 * t;
        float eh = __shfl(ev, f >> 6, 32);
        atomicAdd(&md[f], eh * (float)hs[f]);
    }
}

__global__ void edge_max2(const int* __restrict__ dst, const int* __restrict__ src,
                          const float* __restrict__ pd, const float* __restrict__ ps,
                          unsigned* __restrict__ menc, long long E) {
    long long e = (long long)blockIdx.x * blockDim.x + threadIdx.x;
    if (e >= E) return;
    int d = dst[e], s = src[e];
    float z = pd[d] + ps[s];
    float sc = -(z > 0.f ? z : ALPHA * z);
    atomicMax(&menc[d], enc_f32(sc));
}

__global__ __launch_bounds__(256) void edge_acc2(
    const int* __restrict__ dst, const int* __restrict__ src,
    const float* __restrict__ pd, const float* __restrict__ ps,
    const unsigned* __restrict__ menc, float* __restrict__ den,
    const _Float16* __restrict__ H2, float* __restrict__ msg2, long long E)
{
    const int lane = threadIdx.x & 31;
    long long e = (long long)blockIdx.x * 8 + (threadIdx.x >> 5);
    if (e >= E) return;
    const int d = dst[e], s = src[e];
    float z = pd[d] + ps[s];
    float sc = -(z > 0.f ? z : ALPHA * z);
    float ev = __expf(sc - dec_f32(menc[d]));
    if (lane == 0) atomicAdd(&den[d], ev);
    const _Float16* hs = H2 + (size_t)s * 256;
    float* md = msg2 + (size_t)d * 256;
    #pragma unroll
    for (int t = 0; t < 8; ++t) {
        const int f = lane + 32 * t;
        atomicAdd(&md[f], ev * (float)hs[f]);
    }
}

// x2 = f16(elu(msg/denom[head]))
__global__ void finalize1(const float* __restrict__ msg, const float* __restrict__ den,
                          _Float16* __restrict__ x2, int NV) {
    long long i = (long long)blockIdx.x * blockDim.x + threadIdx.x;   // n*512+f
    if (i >= (long long)NV * 512) return;
    int f = (int)(i & 511); int n = (int)(i >> 9);
    float v = msg[i] / den[(size_t)(f >> 6) * NV + n];
    x2[i] = (_Float16)elu_f(v);
}

// hidden = log_softmax(elu(msg2/denom2)) in place; one wave per row of 256
__global__ __launch_bounds__(256) void finalize2(float* __restrict__ msg2,
                                                 const float* __restrict__ den2, int NV) {
    const int lane = threadIdx.x & 31;
    const int n = blockIdx.x * 8 + (threadIdx.x >> 5);
    if (n >= NV) return;
    float* row = msg2 + (size_t)n * 256;
    const float inv = 1.f / den2[n];
    float vals[8]; float mx = -3.4e38f;
    #pragma unroll
    for (int t = 0; t < 8; ++t) {
        float o = elu_f(row[lane + 32 * t] * inv);
        vals[t] = o; mx = fmaxf(mx, o);
    }
    #pragma unroll
    for (int off = 16; off; off >>= 1) mx = fmaxf(mx, __shfl_xor(mx, off, 32));
    float se = 0.f;
    #pragma unroll
    for (int t = 0; t < 8; ++t) se += __expf(vals[t] - mx);
    #pragma unroll
    for (int off = 16; off; off >>= 1) se += __shfl_xor(se, off, 32);
    const float lse = mx + __logf(se);
    #pragma unroll
    for (int t = 0; t < 8; ++t) row[lane + 32 * t] = vals[t] - lse;
}

__global__ void gather_rows(const float* __restrict__ hidden, const int* __restrict__ ids,
                            _Float16* __restrict__ g, int B) {
    long long i = (long long)blockIdx.x * blockDim.x + threadIdx.x;
    if (i >= (long long)B * 256) return;
    int b = (int)(i >> 8), f = (int)(i & 255);
    g[i] = (_Float16)hidden[(size_t)ids[b] * 256 + f];
}

// ============================ host ============================

static inline long long cdiv_ll(long long a, long long b) { return (a + b - 1) / b; }

extern "C" void kernel_launch(void* const* d_in, const int* in_sizes, int n_in,
                              void* d_out, int out_size, void* d_ws, size_t ws_size,
                              hipStream_t stream)
{
    const float* vertices = (const float*)d_in[0];
    const int*   edges    = (const int*)d_in[1];
    const int*   vids     = (const int*)d_in[2];
    const float* W_heads  = (const float*)d_in[3];
    const float* a_heads  = (const float*)d_in[4];
    const float* W_out    = (const float*)d_in[5];
    const float* a_out    = (const float*)d_in[6];
    const float* W1       = (const float*)d_in[7];
    const float* b1       = (const float*)d_in[8];
    const float* W2       = (const float*)d_in[9];
    const float* b2       = (const float*)d_in[10];

    const int       NV = in_sizes[0] / 256;        // 50000
    const long long E  = in_sizes[1] / 2;          // 1,650,000
    const int       B  = in_sizes[2];              // 4096
    const int* dst = edges;
    const int* src = edges + E;

    char* ws = (char*)d_ws; size_t off = 0;
    auto alloc = [&](size_t bytes) -> char* {
        char* p = ws + off;
        off = (off + bytes + 255) & ~(size_t)255;
        return p;
    };
    _Float16* X16    = (_Float16*)alloc((size_t)NV * 256 * 2);
    _Float16* Wcat16 = (_Float16*)alloc((size_t)256 * 512 * 2);
    _Float16* Wout16 = (_Float16*)alloc((size_t)512 * 256 * 2);
    _Float16* W1_16  = (_Float16*)alloc((size_t)256 * 1024 * 2);
    _Float16* W2_16  = (_Float16*)alloc((size_t)1024 * 256 * 2);
    _Float16* H16    = (_Float16*)alloc((size_t)NV * 512 * 2);
    float*    pd1    = (float*)   alloc((size_t)8 * NV * 4);
    float*    ps1    = (float*)   alloc((size_t)8 * NV * 4);
    unsigned* m1     = (unsigned*)alloc((size_t)8 * NV * 4);
    float*    den1   = (float*)   alloc((size_t)8 * NV * 4);
    float*    msg1   = (float*)   alloc((size_t)NV * 512 * 4);
    _Float16* x2_16  = (_Float16*)alloc((size_t)NV * 512 * 2);
    _Float16* H2_16  = (_Float16*)alloc((size_t)NV * 256 * 2);
    float*    pd2    = (float*)   alloc((size_t)NV * 4);
    float*    ps2    = (float*)   alloc((size_t)NV * 4);
    unsigned* m2     = (unsigned*)alloc((size_t)NV * 4);
    float*    den2   = (float*)   alloc((size_t)NV * 4);
    float*    msg2   = (float*)   alloc((size_t)NV * 256 * 4);  // becomes hidden in-place
    _Float16* g16    = (_Float16*)alloc((size_t)B * 256 * 2);
    _Float16* T16    = (_Float16*)alloc((size_t)B * 1024 * 2);
    (void)ws_size; (void)n_in; (void)out_size;

    const int TB = 256;
    #define GRID1D(n) dim3((unsigned)cdiv_ll((n), TB)), dim3(TB)

    // ---- init accumulators ----
    fill_u32<<<GRID1D(8LL*NV),      0, stream>>>(m1, ENC_NEG_INF, 8LL * NV);
    fill_u32<<<GRID1D(8LL*NV),      0, stream>>>((unsigned*)den1, 0u, 8LL * NV);
    fill_u32<<<GRID1D((long long)NV*512), 0, stream>>>((unsigned*)msg1, 0u, (long long)NV * 512);
    fill_u32<<<GRID1D((long long)NV),     0, stream>>>(m2, ENC_NEG_INF, NV);
    fill_u32<<<GRID1D((long long)NV),     0, stream>>>((unsigned*)den2, 0u, NV);
    fill_u32<<<GRID1D((long long)NV*256), 0, stream>>>((unsigned*)msg2, 0u, (long long)NV * 256);

    // ---- precision conversion (f32 -> f16 GEMM operands) ----
    f32_to_f16_k<<<GRID1D((long long)NV*256), 0, stream>>>(vertices, X16, (long long)NV * 256);
    make_wcat_k <<<GRID1D(8LL*256*64),        0, stream>>>(W_heads, Wcat16);
    f32_to_f16_k<<<GRID1D(512LL*256),         0, stream>>>(W_out, Wout16, 512LL * 256);
    f32_to_f16_k<<<GRID1D(256LL*1024),        0, stream>>>(W1, W1_16, 256LL * 1024);
    f32_to_f16_k<<<GRID1D(1024LL*256),        0, stream>>>(W2, W2_16, 1024LL * 256);

    // ---- layer 1: H = X @ Wcat   [NV,512] ----
    gemm_wmma_f16<1,0,0><<<dim3(512/64, (NV+31)/32), dim3(256), 0, stream>>>(
        X16, Wcat16, H16, nullptr, NV, 512, 256);
    head_proj1<<<GRID1D(8LL*NV), 0, stream>>>(H16, a_heads, pd1, ps1, NV);
    edge_max1<<<GRID1D(E*8), 0, stream>>>(dst, src, pd1, ps1, m1, E, NV);
    edge_acc1<<<dim3((unsigned)cdiv_ll(E,8)), dim3(256), 0, stream>>>(
        dst, src, pd1, ps1, m1, den1, H16, msg1, E, NV);
    finalize1<<<GRID1D((long long)NV*512), 0, stream>>>(msg1, den1, x2_16, NV);

    // ---- layer 2: H2 = x2 @ W_out   [NV,256] ----
    gemm_wmma_f16<1,0,0><<<dim3(256/64, (NV+31)/32), dim3(256), 0, stream>>>(
        x2_16, Wout16, H2_16, nullptr, NV, 256, 512);
    head_proj2<<<GRID1D((long long)NV), 0, stream>>>(H2_16, a_out, pd2, ps2, NV);
    edge_max2<<<GRID1D(E), 0, stream>>>(dst, src, pd2, ps2, m2, E);
    edge_acc2<<<dim3((unsigned)cdiv_ll(E,8)), dim3(256), 0, stream>>>(
        dst, src, pd2, ps2, m2, den2, H2_16, msg2, E);
    finalize2<<<dim3((NV+7)/8), dim3(256), 0, stream>>>(msg2, den2, NV);

    // ---- gather + MLP ----
    gather_rows<<<GRID1D((long long)B*256), 0, stream>>>(msg2, vids, g16, B);
    gemm_wmma_f16<1,1,1><<<dim3(1024/64, (B+31)/32), dim3(256), 0, stream>>>(
        g16, W1_16, T16, b1, B, 1024, 256);
    gemm_wmma_f16<0,1,0><<<dim3(256/64, (B+31)/32), dim3(256), 0, stream>>>(
        T16, W2_16, d_out, b2, B, 256, 1024);
    #undef GRID1D
}
